// SAGNetworkHierarchical_34076270526879
// MI455X (gfx1250) — matrix operand order repaired and checked by reading, not compiled, not run on hardware
//
#include <hip/hip_runtime.h>
#include <math.h>

typedef __attribute__((ext_vector_type(16))) _Float16 v16h;
typedef __attribute__((ext_vector_type(8)))  _Float16 v8h;
typedef __attribute__((ext_vector_type(8)))  float    v8f;

#define DEVFN static __device__ __forceinline__

// ---------------------------------------------------------------------------
// WMMA fragment loaders (v_wmma_f32_16x16x32_f16, this file's contiguous-k order, wave32)
// A (16x32, row-major rows): lane<16 holds K k0+0..15, lane>=16 k0+16..31
// B supplied as Bt (row-major [N][K]): lane<16 holds K k0+0..15, lane>=16 k0+16..31
// ---------------------------------------------------------------------------
DEVFN v16h load_a_frag(const _Float16* __restrict__ row, int k0, int half) {
    return *(const v16h*)(row + k0 + 16 * half);
}
DEVFN v16h load_bt_frag(const _Float16* __restrict__ row, int k0, int half) {
    return *(const v16h*)(row + k0 + 16 * half);
}

// ---------------------------------------------------------------------------
// GEMM-XW: xwT[b][o][j] = sum_f xs[b][j][f] * Wt[o][f]    (f16 in, f16 out)
// Each wave: one 16(M=j) x 64(N=o) strip. Transposed C store is contiguous.
// ---------------------------------------------------------------------------
__global__ void __launch_bounds__(256) gemm_xw_wmma(
    const _Float16* __restrict__ xs, const _Float16* __restrict__ wt,
    _Float16* __restrict__ xwT, int M, int K, int N)
{
    int w    = (blockIdx.x * blockDim.x + threadIdx.x) >> 5;
    int lane = threadIdx.x & 31;
    int half = lane >> 4, ln = lane & 15;
    int groups = N >> 6;          // N/64
    int tilesM = M >> 4;          // M/16
    int wpb = tilesM * groups;
    int b  = w / wpb;
    int r  = w - b * wpb;
    int mt = r / groups;
    int ng = r - mt * groups;
    int m0 = mt << 4, n0 = ng << 6;

    const _Float16* arow = xs + ((size_t)b * M + (size_t)(m0 + ln)) * K;
    v8f zero = {};
    v8f acc[4] = {zero, zero, zero, zero};
    for (int k0 = 0; k0 < K; k0 += 32) {
        __builtin_prefetch(arow + k0 + 128, 0, 1);
        v16h a = load_a_frag(arow, k0, half);
#pragma unroll
        for (int t = 0; t < 4; t++) {
            v16h bf = load_bt_frag(wt + (size_t)(n0 + t * 16 + ln) * K, k0, half);
            acc[t] = __builtin_amdgcn_wmma_f32_16x16x32_f16(
                false, a, false, bf, (short)0, acc[t], false, false);
        }
    }
    _Float16* outb = xwT + (size_t)b * N * M;
#pragma unroll
    for (int t = 0; t < 4; t++) {
        int n = n0 + t * 16 + ln;
        v8h o;
#pragma unroll
        for (int i = 0; i < 8; i++) o[i] = (_Float16)acc[t][i];
        *(v8h*)(outb + (size_t)n * M + m0 + 8 * half) = o;   // contiguous 16B
    }
}

// ---------------------------------------------------------------------------
// GEMM-AGG: h[b][i][o] = relu( (sum_j At[b][i][j]*xwT[b][o][j]) * dri[b][i] + bias[o] )
// ---------------------------------------------------------------------------
__global__ void __launch_bounds__(256) gemm_agg_wmma(
    const _Float16* __restrict__ At, const _Float16* __restrict__ xwT,
    const float* __restrict__ dri, const float* __restrict__ bias,
    float* __restrict__ hout, int M, int K, int N)
{
    int w    = (blockIdx.x * blockDim.x + threadIdx.x) >> 5;
    int lane = threadIdx.x & 31;
    int half = lane >> 4, ln = lane & 15;
    int groups = N >> 6;
    int tilesM = M >> 4;
    int wpb = tilesM * groups;
    int b  = w / wpb;
    int r  = w - b * wpb;
    int mt = r / groups;
    int ng = r - mt * groups;
    int m0 = mt << 4, n0 = ng << 6;

    const _Float16* arow = At + ((size_t)b * M + (size_t)(m0 + ln)) * K;
    const _Float16* bb   = xwT + (size_t)b * N * K;
    v8f zero = {};
    v8f acc[4] = {zero, zero, zero, zero};
    for (int k0 = 0; k0 < K; k0 += 32) {
        __builtin_prefetch(arow + k0 + 128, 0, 1);
        v16h a = load_a_frag(arow, k0, half);
#pragma unroll
        for (int t = 0; t < 4; t++) {
            v16h bf = load_bt_frag(bb + (size_t)(n0 + t * 16 + ln) * K, k0, half);
            acc[t] = __builtin_amdgcn_wmma_f32_16x16x32_f16(
                false, a, false, bf, (short)0, acc[t], false, false);
        }
    }
    float* outb = hout + (size_t)b * M * N;
#pragma unroll
    for (int t = 0; t < 4; t++) {
        int n = n0 + t * 16 + ln;
        float bv = bias[n];
#pragma unroll
        for (int v = 0; v < 8; v++) {
            int m = m0 + v + 8 * half;
            float val = acc[t][v] * dri[b * M + m] + bv;
            outb[(size_t)m * N + n] = val > 0.f ? val : 0.f;
        }
    }
}

// ---------------------------------------------------------------------------
// Elementwise / reduction helpers (f32 precision, trivially memory-cheap)
// ---------------------------------------------------------------------------
__global__ void fill0_kernel(float* p, int total) {
    int t = blockIdx.x * blockDim.x + threadIdx.x;
    if (t < total) p[t] = 0.f;
}

__global__ void degree_row_kernel(const float* __restrict__ A, float* dro, int n, int total) {
    int t = blockIdx.x * blockDim.x + threadIdx.x;
    if (t >= total) return;
    int b = t / n, j = t - b * n;
    const float* row = A + ((size_t)b * n + j) * n;
    float s = 0.f;
    for (int i = 0; i < n; i++) s += row[i];
    dro[t] = rsqrtf(fmaxf(s, 1.f));
}

__global__ void degree_col_kernel(const float* __restrict__ A, float* dri, int n, int total) {
    int t = blockIdx.x * blockDim.x + threadIdx.x;
    if (t >= total) return;
    int b = t / n, i = t - b * n;
    const float* p = A + (size_t)b * n * n + i;
    float s = 0.f;
    for (int j = 0; j < n; j++) s += p[(size_t)j * n];
    dri[t] = rsqrtf(fmaxf(s, 1.f));
}

__global__ void scale_convert_kernel(const float* __restrict__ h, const float* __restrict__ dro,
                                     _Float16* __restrict__ xs, int F, int total) {
    int t = blockIdx.x * blockDim.x + threadIdx.x;
    if (t >= total) return;
    xs[t] = (_Float16)(h[t] * dro[t / F]);
}

__global__ void wT_convert_kernel(const float* __restrict__ W, _Float16* __restrict__ Wt,
                                  int K, int N, int total) {
    int t = blockIdx.x * blockDim.x + threadIdx.x;
    if (t >= total) return;
    int o = t / K, f = t - o * K;
    Wt[t] = (_Float16)W[(size_t)f * N + o];
}

// Tiled transpose + f32->f16: At[b][i][j] = A[b][j][i]
__global__ void at_convert_kernel(const float* __restrict__ A, _Float16* __restrict__ At, int n) {
    __shared__ float tile[32][33];
    int b = blockIdx.z;
    int i0 = blockIdx.x * 32, j0 = blockIdx.y * 32;
    const float* Ab = A + (size_t)b * n * n;
    _Float16* Atb = At + (size_t)b * n * n;
    int tx = threadIdx.x, ty = threadIdx.y;
#pragma unroll
    for (int yy = 0; yy < 32; yy += 8)
        tile[ty + yy][tx] = Ab[(size_t)(j0 + ty + yy) * n + i0 + tx];
    __syncthreads();
#pragma unroll
    for (int yy = 0; yy < 32; yy += 8)
        Atb[(size_t)(i0 + ty + yy) * n + j0 + tx] = (_Float16)tile[tx][ty + yy];
}

// score phase 1: t[b,j] = (h3[b,j,:].(Ws1+Ws2)) * dro[b,j]   (one wave per row)
__global__ void score1_kernel(const float* __restrict__ h, const float* __restrict__ dro,
                              const float* __restrict__ ws1, const float* __restrict__ ws2,
                              float* __restrict__ tout, int n, int F) {
    int w = (blockIdx.x * blockDim.x + threadIdx.x) >> 5;
    int lane = threadIdx.x & 31;
    const float* row = h + (size_t)w * F;
    float acc = 0.f;
    for (int f = lane; f < F; f += 32) acc += row[f] * (ws1[f] + ws2[f]);
#pragma unroll
    for (int off = 16; off > 0; off >>= 1) acc += __shfl_xor(acc, off, 32);
    if (lane == 0) tout[w] = acc * dro[w];
}

// score phase 2: s[b,i] = 0.5*((sum_j A[b,j,i]*t[b,j])*dri + bs1 + bs2)
__global__ void score2_kernel(const float* __restrict__ A, const float* __restrict__ tin,
                              const float* __restrict__ dri,
                              const float* __restrict__ bs1, const float* __restrict__ bs2,
                              float* __restrict__ s, int n, int total) {
    int t = blockIdx.x * blockDim.x + threadIdx.x;
    if (t >= total) return;
    int b = t / n, i = t - b * n;
    const float* col = A + (size_t)b * n * n + i;
    const float* tv = tin + b * n;
    float acc = 0.f;
    for (int j = 0; j < n; j++) acc += col[(size_t)j * n] * tv[j];
    s[t] = 0.5f * (acc * dri[t] + bs1[0] + bs2[0]);
}

// top-k, descending, stable ties (lowest index first). One block per batch.
__global__ void topk_kernel(const float* __restrict__ s, int* __restrict__ idx, int n, int k) {
    __shared__ float sv[512];
    __shared__ float rv[256];
    __shared__ int   ri[256];
    int b = blockIdx.x, tid = threadIdx.x;
    for (int i = tid; i < n; i += 256) sv[i] = s[b * n + i];
    __syncthreads();
    for (int r = 0; r < k; r++) {
        float best = -3.4e38f; int bi = n;
        for (int i = tid; i < n; i += 256) {
            float v = sv[i];
            if (v > best || (v == best && i < bi)) { best = v; bi = i; }
        }
        rv[tid] = best; ri[tid] = bi;
        __syncthreads();
        for (int off = 128; off > 0; off >>= 1) {
            if (tid < off) {
                if (rv[tid + off] > rv[tid] ||
                    (rv[tid + off] == rv[tid] && ri[tid + off] < ri[tid])) {
                    rv[tid] = rv[tid + off]; ri[tid] = ri[tid + off];
                }
            }
            __syncthreads();
        }
        if (tid == 0) { idx[b * k + r] = ri[0]; sv[ri[0]] = -3.4e38f; }
        __syncthreads();
    }
}

__global__ void gather_h_kernel(const float* __restrict__ h, const float* __restrict__ s,
                                const int* __restrict__ idx, float* __restrict__ hp,
                                int n, int k, int F, int total) {
    int t = blockIdx.x * blockDim.x + threadIdx.x;
    if (t >= total) return;
    int f = t % F;
    int br = t / F;
    int b = br / k, r = br - b * k;
    int src = idx[b * k + r];
    hp[t] = h[((size_t)b * n + src) * F + f] * tanhf(s[b * n + src]);
}

__global__ void gather_A_kernel(const float* __restrict__ A, const int* __restrict__ idx,
                                float* __restrict__ Ap, int n, int k, int total) {
    int t = blockIdx.x * blockDim.x + threadIdx.x;
    if (t >= total) return;
    int c = t % k;
    int br = t / k;
    int b = br / k, r = br - b * k;
    Ap[t] = A[(size_t)b * n * n + (size_t)idx[b * k + r] * n + idx[b * k + c]];
}

__global__ void maxpool_acc_kernel(const float* __restrict__ hp, float* __restrict__ readout,
                                   int k, int F, int total) {
    int t = blockIdx.x * blockDim.x + threadIdx.x;
    if (t >= total) return;
    int b = t / F, f = t - b * F;
    const float* p = hp + (size_t)b * k * F + f;
    float m = -3.4e38f;
    for (int r = 0; r < k; r++) m = fmaxf(m, p[(size_t)r * F]);
    readout[(size_t)b * 2 * F + f]     += m;
    readout[(size_t)b * 2 * F + F + f] += m;
}

__global__ void linear_kernel(const float* __restrict__ in, const float* __restrict__ W,
                              const float* __restrict__ bias, float* __restrict__ out,
                              int K, int N, int doRelu, int total) {
    int t = blockIdx.x * blockDim.x + threadIdx.x;
    if (t >= total) return;
    int b = t / N, o = t - b * N;
    const float* row = in + (size_t)b * K;
    float acc = bias[o];
    for (int kk = 0; kk < K; kk++) acc += row[kk] * W[(size_t)kk * N + o];
    out[t] = doRelu ? fmaxf(acc, 0.f) : acc;
}

// ---------------------------------------------------------------------------
// Host orchestration
// ---------------------------------------------------------------------------
static inline unsigned cdiv(long long a, int b) { return (unsigned)((a + b - 1) / b); }

extern "C" void kernel_launch(void* const* d_in, const int* in_sizes, int n_in,
                              void* d_out, int out_size, void* d_ws, size_t ws_size,
                              hipStream_t stream) {
    (void)in_sizes; (void)n_in; (void)out_size; (void)ws_size;
    const int B = 64, N0 = 512, F0 = 1280, H = 512, OUT = 1000;

    const float* x   = (const float*)d_in[0];
    const float* adj = (const float*)d_in[1];
    const float* Wc1[3] = {(const float*)d_in[2],  (const float*)d_in[10], (const float*)d_in[18]};
    const float* bc1[3] = {(const float*)d_in[3],  (const float*)d_in[11], (const float*)d_in[19]};
    const float* Wc2[3] = {(const float*)d_in[4],  (const float*)d_in[12], (const float*)d_in[20]};
    const float* bc2[3] = {(const float*)d_in[5],  (const float*)d_in[13], (const float*)d_in[21]};
    const float* Ws1[3] = {(const float*)d_in[6],  (const float*)d_in[14], (const float*)d_in[22]};
    const float* bs1[3] = {(const float*)d_in[7],  (const float*)d_in[15], (const float*)d_in[23]};
    const float* Ws2[3] = {(const float*)d_in[8],  (const float*)d_in[16], (const float*)d_in[24]};
    const float* bs2[3] = {(const float*)d_in[9],  (const float*)d_in[17], (const float*)d_in[25]};
    const float* l1w = (const float*)d_in[26]; const float* l1b = (const float*)d_in[27];
    const float* l2w = (const float*)d_in[28]; const float* l2b = (const float*)d_in[29];
    const float* l3w = (const float*)d_in[30]; const float* l3b = (const float*)d_in[31];
    float* out = (float*)d_out;

    // workspace bump allocator (256B aligned)
    char* base = (char*)d_ws; size_t off = 0;
    auto alloc = [&](size_t bytes) -> char* {
        char* r = base + off;
        off = (off + bytes + 255) & ~(size_t)255;
        return r;
    };
    _Float16* At  = (_Float16*)alloc((size_t)B * N0 * N0 * 2);   // 32 MB
    _Float16* xs  = (_Float16*)alloc((size_t)B * N0 * F0 * 2);   // 80 MB
    _Float16* xwT = (_Float16*)alloc((size_t)B * N0 * H  * 2);   // 32 MB
    _Float16* Wt  = (_Float16*)alloc((size_t)F0 * H * 2);
    float* hA  = (float*)alloc((size_t)B * N0 * H * 4);          // 64 MB
    float* hB  = (float*)alloc((size_t)B * 256 * H * 4);         // 32 MB
    float* Ap1 = (float*)alloc((size_t)B * 256 * 256 * 4);       // 16 MB
    float* Ap2 = (float*)alloc((size_t)B * 128 * 128 * 4);       //  4 MB
    float* dro = (float*)alloc((size_t)B * N0 * 4);
    float* dri = (float*)alloc((size_t)B * N0 * 4);
    float* tb  = (float*)alloc((size_t)B * N0 * 4);
    float* sb  = (float*)alloc((size_t)B * N0 * 4);
    int*   idx = (int*)  alloc((size_t)B * 256 * 4);
    float* readout = (float*)alloc((size_t)B * 2 * H * 4);
    float* m1  = (float*)alloc((size_t)B * H * 4);
    float* m2  = (float*)alloc((size_t)B * H * 4);

    fill0_kernel<<<cdiv(B * 2 * H, 256), 256, 0, stream>>>(readout, B * 2 * H);

    const int   ns[3]  = {512, 256, 128};
    float*      hw[3]  = {hA, hB, hA};   // in-place gconv working buffers
    float*      hpw[3] = {hB, hA, hB};   // pooled outputs
    float*      An[3]  = {Ap1, Ap2, nullptr};

    const float* Acur = adj;
    const float* hin  = x;
    int Fin = F0;

    for (int blk = 0; blk < 3; blk++) {
        int n = ns[blk], k = n / 2;
        long long bn = (long long)B * n;

        degree_row_kernel<<<cdiv(bn, 256), 256, 0, stream>>>(Acur, dro, n, (int)bn);
        degree_col_kernel<<<cdiv(bn, 256), 256, 0, stream>>>(Acur, dri, n, (int)bn);
        at_convert_kernel<<<dim3(n / 32, n / 32, B), dim3(32, 8), 0, stream>>>(Acur, At, n);

        // gconv helper: conv with weights W/bias into hw[blk] (xs from hsrc)
        auto gconv = [&](const float* hsrc, int Fk, const float* Wg, const float* bg) {
            long long tot = bn * Fk;
            wT_convert_kernel<<<cdiv((long long)Fk * H, 256), 256, 0, stream>>>(
                Wg, Wt, Fk, H, Fk * H);
            scale_convert_kernel<<<cdiv(tot, 256), 256, 0, stream>>>(
                hsrc, dro, xs, Fk, (int)tot);
            unsigned gx = (unsigned)((long long)B * (n / 16) * (H / 64) / 8);
            gemm_xw_wmma<<<gx, 256, 0, stream>>>(xs, Wt, xwT, n, Fk, H);
            gemm_agg_wmma<<<gx, 256, 0, stream>>>(At, xwT, dri, bg, hw[blk], n, n, H);
        };

        gconv(hin,     Fin, Wc1[blk], bc1[blk]);   // conv1 + relu
        gconv(hw[blk], H,   Wc2[blk], bc2[blk]);   // conv2 + relu (shared W, twice)
        gconv(hw[blk], H,   Wc2[blk], bc2[blk]);

        // SAGPool scores
        score1_kernel<<<cdiv(bn * 32, 256), 256, 0, stream>>>(
            hw[blk], dro, Ws1[blk], Ws2[blk], tb, n, H);
        score2_kernel<<<cdiv(bn, 256), 256, 0, stream>>>(
            Acur, tb, dri, bs1[blk], bs2[blk], sb, n, (int)bn);
        topk_kernel<<<B, 256, 0, stream>>>(sb, idx, n, k);

        long long thp = (long long)B * k * H;
        gather_h_kernel<<<cdiv(thp, 256), 256, 0, stream>>>(
            hw[blk], sb, idx, hpw[blk], n, k, H, (int)thp);
        maxpool_acc_kernel<<<cdiv((long long)B * H, 256), 256, 0, stream>>>(
            hpw[blk], readout, k, H, B * H);
        if (blk < 2) {
            long long ta = (long long)B * k * k;
            gather_A_kernel<<<cdiv(ta, 256), 256, 0, stream>>>(
                Acur, idx, An[blk], n, k, (int)ta);
            Acur = An[blk];
        }
        hin = hpw[blk];
        Fin = H;
    }

    // MLP head (tiny; plain f32)
    linear_kernel<<<cdiv(B * H, 256), 256, 0, stream>>>(readout, l1w, l1b, m1, 2 * H, H, 1, B * H);
    linear_kernel<<<cdiv(B * H, 256), 256, 0, stream>>>(m1, l2w, l2b, m2, H, H, 1, B * H);
    linear_kernel<<<cdiv(B * OUT, 256), 256, 0, stream>>>(m2, l3w, l3b, out, H, OUT, 0, B * OUT);
}
